// PointNetSAModule_47571057771109
// MI455X (gfx1250) — matrix operand order repaired and checked by reading, not compiled, not run on hardware
//
#include <hip/hip_runtime.h>
#include <hip/hip_bf16.h>

#define Bn   8
#define Np   8192
#define Mc   1024
#define Kn   64
#define CIN  64
#define CH0  67      // 3 + 64 input channels
#define CH0P 96      // padded to multiple of 32
#define ST0  104     // LDS stride (halves) for staged input / layer2 output
#define STY  72      // LDS stride (halves) for layer1 output
#define R2   0.04f   // radius^2

typedef _Float16 h8v  __attribute__((ext_vector_type(8)));
typedef _Float16 v16h __attribute__((ext_vector_type(16)));
typedef float    v8f  __attribute__((ext_vector_type(8)));

union HV { v16h v; h8v h[2]; };

// ---------------------------------------------------------------------------
// Kernel 1: farthest point sampling. One 1024-thread block per batch.
// Points + running min-dists live in registers; argmax via shfl + tiny LDS.
// ---------------------------------------------------------------------------
__global__ __launch_bounds__(1024) void fps_kernel(const float* __restrict__ xyz,
                                                   int* __restrict__ cidx) {
  const int b = blockIdx.x;
  const int t = threadIdx.x;
  const float* xb = xyz + (size_t)b * 3 * Np;

  float px[8], py[8], pz[8], dm[8];
#pragma unroll
  for (int e = 0; e < 8; ++e) {
    int j = e * 1024 + t;
    px[e] = xb[j];
    py[e] = xb[Np + j];
    pz[e] = xb[2 * Np + j];
    dm[e] = 1e10f;
  }

  __shared__ float lx, ly, lz;
  __shared__ float wval[32];
  __shared__ int   widx[32];
  __shared__ int   sidx;

  if (t == 0) { cidx[b * Mc] = 0; lx = xb[0]; ly = xb[Np]; lz = xb[2 * Np]; }
  __syncthreads();

  for (int s = 1; s < Mc; ++s) {
    float cx = lx, cy = ly, cz = lz;
    float bv = -1.0f; int bi = 0;
#pragma unroll
    for (int e = 0; e < 8; ++e) {
      float dx = px[e] - cx, dy = py[e] - cy, dz = pz[e] - cz;
      float d = dx * dx + dy * dy + dz * dz;
      dm[e] = fminf(dm[e], d);
      int j = e * 1024 + t;
      if (dm[e] > bv || (dm[e] == bv && j < bi)) { bv = dm[e]; bi = j; }
    }
    // wave32 reduction (argmax, first-index tie-break)
#pragma unroll
    for (int off = 16; off >= 1; off >>= 1) {
      float ov = __shfl_xor(bv, off);
      int   oi = __shfl_xor(bi, off);
      if (ov > bv || (ov == bv && oi < bi)) { bv = ov; bi = oi; }
    }
    int lane = t & 31, wv = t >> 5;
    if (lane == 0) { wval[wv] = bv; widx[wv] = bi; }
    __syncthreads();
    if (wv == 0) {
      bv = wval[lane]; bi = widx[lane];
#pragma unroll
      for (int off = 16; off >= 1; off >>= 1) {
        float ov = __shfl_xor(bv, off);
        int   oi = __shfl_xor(bi, off);
        if (ov > bv || (ov == bv && oi < bi)) { bv = ov; bi = oi; }
      }
      if (lane == 0) { cidx[b * Mc + s] = bi; sidx = bi; }
    }
    __syncthreads();
    int own = sidx;
    if (t == (own & 1023)) { int e = own >> 10; lx = px[e]; ly = py[e]; lz = pz[e]; }
    __syncthreads();
  }
}

// ---------------------------------------------------------------------------
// Kernel 2: gather centroid coordinates -> d_out[0 : B*3*M]
// ---------------------------------------------------------------------------
__global__ __launch_bounds__(256) void gather_newxyz_kernel(const float* __restrict__ xyz,
                                                            const int* __restrict__ cidx,
                                                            float* __restrict__ out) {
  int g = blockIdx.x * 256 + threadIdx.x;      // 0 .. B*M-1
  int b = g / Mc, m = g % Mc;
  int idx = cidx[g];
  const float* xb = xyz + (size_t)b * 3 * Np;
  float* ob = out + (size_t)b * 3 * Mc;
  ob[m]          = xb[idx];
  ob[Mc + m]     = xb[Np + idx];
  ob[2 * Mc + m] = xb[2 * Np + idx];
}

// ---------------------------------------------------------------------------
// Kernel 3: ball query. One wave per centroid; ballot + prefix popc keeps
// neighbor indices in ascending order (reference's stable-sort semantics),
// padded with the first hit (or 0 if no hit).
// ---------------------------------------------------------------------------
__global__ __launch_bounds__(256) void ballq_kernel(const float* __restrict__ xyz,
                                                    const float* __restrict__ newxyz,
                                                    int* __restrict__ nidx) {
  int wv = threadIdx.x >> 5, lane = threadIdx.x & 31;
  int g = blockIdx.x * 8 + wv;                 // centroid id
  int b = g / Mc, m = g % Mc;
  float cx = newxyz[(size_t)(b * 3 + 0) * Mc + m];
  float cy = newxyz[(size_t)(b * 3 + 1) * Mc + m];
  float cz = newxyz[(size_t)(b * 3 + 2) * Mc + m];
  const float* xb = xyz + (size_t)b * 3 * Np;
  int* outp = nidx + (size_t)g * Kn;

  int cnt = 0, firstIdx = 0;
  for (int c0 = 0; c0 < Np; c0 += 32) {
    int j = c0 + lane;
    float dx = xb[j] - cx, dy = xb[Np + j] - cy, dz = xb[2 * Np + j] - cz;
    float d2 = dx * dx + dy * dy + dz * dz;
    bool inr = d2 < R2;
    unsigned mask = (unsigned)__ballot(inr);
    if (mask) {
      if (cnt == 0) firstIdx = c0 + __ffs(mask) - 1;
      int pos = cnt + __popc(mask & ((1u << lane) - 1u));
      if (inr && pos < Kn) outp[pos] = j;
      cnt += __popc(mask);
      if (cnt >= Kn) break;
    }
  }
  if (cnt > Kn) cnt = Kn;
  int pad = (cnt > 0) ? firstIdx : 0;
  for (int p = cnt + lane; p < Kn; p += 32) outp[p] = pad;
}

// ---------------------------------------------------------------------------
// WMMA fragment helpers (ISA 16-bit A / B layouts, wave32)
// ---------------------------------------------------------------------------

// A (weights) fragment. The empty asm redefines the row offset every call so
// LICM cannot hoist the 36 weight fragments out of the n-tile loop (which
// previously blew past 256 VGPRs and forced s_set_vgpr_msb addressing).
__device__ inline v16h load_afrag(const _Float16* __restrict__ w, int stride,
                                  int obase, int ib, int lane) {
  int o = obase + (lane & 15);
  asm volatile("" : "+v"(o));                  // defeat LICM, keep load in-loop
  int off = (lane < 16) ? 0 : 8;               // lanes 0-15: K {0-7,16-23}; 16-31: {8-15,24-31}
  HV u;
  u.h[0] = *(const h8v*)(w + o * stride + ib + off);
  u.h[1] = *(const h8v*)(w + o * stride + ib + 16 + off);
  return u.v;
}

__device__ inline v16h load_bfrag(const _Float16* __restrict__ x, int stride,
                                  int ib, int lane) {
  int n = lane & 15;
  int kh = (lane < 16) ? 0 : 8;                // lanes 0-15: K {0-7,16-23}; 16-31: {8-15,24-31}
  HV u;
  u.h[0] = *(const h8v*)(x + n * stride + ib + kh);
  u.h[1] = *(const h8v*)(x + n * stride + ib + kh + 16);
  return u.v;
}

// Apply affine+ReLU to a 16x16 f32 accumulator and store f16 [n][o] to LDS.
__device__ inline void store_yfrag(_Float16* __restrict__ y, int stride, int obase,
                                   v8f acc, const float* __restrict__ sc,
                                   const float* __restrict__ bi, int lane) {
  int n = lane & 15;
  int oo = obase + ((lane < 16) ? 0 : 8);      // D: lanes 0-15 -> M r, 16-31 -> M r+8
  asm volatile("" : "+v"(oo));                 // keep scale/bias reads in-loop
  h8v h;
#pragma unroll
  for (int r = 0; r < 8; ++r) {
    float v = acc[r] * sc[oo + r] + bi[oo + r];
    v = v > 0.0f ? v : 0.0f;
    h[r] = (_Float16)v;
  }
  *(h8v*)(y + n * stride + oo) = h;
}

__device__ inline v8f wmma_f16(v16h a, v16h b, v8f c) {
  return __builtin_amdgcn_wmma_f32_16x16x32_f16(false, a, false, b, (short)0, c,
                                                false, false);
}

// ---------------------------------------------------------------------------
// Kernel 4: gather + 3-layer MLP (f16 WMMA, f32 accum) + max over neighbors.
// 128 threads = 4 waves; one wave owns one centroid; n-tiles of 16 neighbors.
// ---------------------------------------------------------------------------
__global__ __launch_bounds__(128) void mlp_kernel(const float* __restrict__ xyz,
                                                  const float* __restrict__ feat,
                                                  const float* __restrict__ newxyz,
                                                  const int* __restrict__ nidx,
                                                  const float* __restrict__ w0,
                                                  const float* __restrict__ s0,
                                                  const float* __restrict__ b0,
                                                  const float* __restrict__ w1,
                                                  const float* __restrict__ s1,
                                                  const float* __restrict__ b1,
                                                  const float* __restrict__ w2,
                                                  const float* __restrict__ s2,
                                                  const float* __restrict__ b2,
                                                  float* __restrict__ outF) {
  __shared__ __align__(16) _Float16 wA0[64 * CH0P];   // w0 padded [o][i], i<67 valid
  __shared__ __align__(16) _Float16 wA1[64 * 64];
  __shared__ __align__(16) _Float16 wA2[128 * 64];
  __shared__ float sc0[64], bi0[64], sc1[64], bi1[64], sc2[128], bi2[128];
  __shared__ __align__(16) _Float16 Xt[4][16 * ST0];  // staged input / layer2 out
  __shared__ __align__(16) _Float16 Yt[4][16 * STY];  // layer1 out
  __shared__ int   nidxS[4][64];
  __shared__ float cen[4][4];

  const int tid = threadIdx.x;
  const int wv = tid >> 5, lane = tid & 31;

  // ---- stage weights (f16) + affine params, cooperatively ----
  for (int e = tid; e < 64 * CH0P; e += 128) {
    int o = e / CH0P, i = e % CH0P;
    wA0[e] = (_Float16)((i < CH0) ? w0[o * CH0 + i] : 0.0f);
  }
  for (int e = tid; e < 64 * 64; e += 128)  wA1[e] = (_Float16)w1[e];
  for (int e = tid; e < 128 * 64; e += 128) wA2[e] = (_Float16)w2[e];
  if (tid < 64)  { sc0[tid] = s0[tid]; bi0[tid] = b0[tid];
                   sc1[tid] = s1[tid]; bi1[tid] = b1[tid]; }
  if (tid < 128) { sc2[tid] = s2[tid]; bi2[tid] = b2[tid]; }
  __syncthreads();

  const int g = blockIdx.x * 4 + wv;           // centroid id
  const int b = g / Mc, m = g % Mc;
  if (lane < 3) cen[wv][lane] = newxyz[(size_t)(b * 3 + lane) * Mc + m];

  const float* xb = xyz  + (size_t)b * 3 * Np;
  const float* fb = feat + (size_t)b * CIN * Np;
  const int*   ng = nidx + (size_t)g * Kn;

  // all 64 neighbor indices, once
  nidxS[wv][lane]      = ng[lane];
  nidxS[wv][32 + lane] = ng[32 + lane];

  v8f zero = {};
  v8f rmax[8];
#pragma unroll
  for (int mt = 0; mt < 8; ++mt) rmax[mt] = zero;   // ReLU outputs >= 0

  _Float16* xt = Xt[wv];
  _Float16* yt = Yt[wv];

  for (int nt = 0; nt < 4; ++nt) {
    // ---- gather 16 neighbors x 96 channels into LDS (f16, [n][c]) ----
    int n = lane & 15;
    int ch = lane >> 4;
    int idx = nidxS[wv][nt * 16 + n];
#pragma unroll 8
    for (int it = 0; it < 48; ++it) {
      int c = it * 2 + ch;
      float val;
      if (c < 3)          val = xb[(size_t)c * Np + idx] - cen[wv][c];
      else if (c < CH0)   val = fb[(size_t)(c - 3) * Np + idx];
      else                val = 0.0f;
      xt[n * ST0 + c] = (_Float16)val;
    }

    // ---- layer 1: 64x16 = w0[64x96] * X[96x16] ----
    {
      v16h bf[3];
#pragma unroll
      for (int ks = 0; ks < 3; ++ks) bf[ks] = load_bfrag(xt, ST0, ks * 32, lane);
#pragma unroll
      for (int mt = 0; mt < 4; ++mt) {
        v8f acc = zero;
#pragma unroll
        for (int ks = 0; ks < 3; ++ks)
          acc = wmma_f16(load_afrag(wA0, CH0P, mt * 16, ks * 32, lane), bf[ks], acc);
        store_yfrag(yt, STY, mt * 16, acc, sc0, bi0, lane);
      }
    }

    // ---- layer 2: 64x16 = w1[64x64] * Y1[64x16]  (writes back into Xt) ----
    {
      v16h bf[2];
#pragma unroll
      for (int ks = 0; ks < 2; ++ks) bf[ks] = load_bfrag(yt, STY, ks * 32, lane);
#pragma unroll
      for (int mt = 0; mt < 4; ++mt) {
        v8f acc = zero;
#pragma unroll
        for (int ks = 0; ks < 2; ++ks)
          acc = wmma_f16(load_afrag(wA1, 64, mt * 16, ks * 32, lane), bf[ks], acc);
        store_yfrag(xt, ST0, mt * 16, acc, sc1, bi1, lane);
      }
    }

    // ---- layer 3: 128x16 = w2[128x64] * Y2[64x16]; fold into running max ----
    {
      v16h bf[2];
#pragma unroll
      for (int ks = 0; ks < 2; ++ks) bf[ks] = load_bfrag(xt, ST0, ks * 32, lane);
#pragma unroll
      for (int mt = 0; mt < 8; ++mt) {
        v8f acc = zero;
#pragma unroll
        for (int ks = 0; ks < 2; ++ks)
          acc = wmma_f16(load_afrag(wA2, 64, mt * 16, ks * 32, lane), bf[ks], acc);
        int oo = mt * 16 + ((lane < 16) ? 0 : 8);
        asm volatile("" : "+v"(oo));           // keep scale/bias reads in-loop
#pragma unroll
        for (int r = 0; r < 8; ++r) {
          float v = acc[r] * sc2[oo + r] + bi2[oo + r];
          v = v > 0.0f ? v : 0.0f;
          rmax[mt][r] = fmaxf(rmax[mt][r], v);
        }
      }
    }
  }

  // ---- max over the 16-lane halves (N dim), then write [B,128,M] ----
#pragma unroll
  for (int mt = 0; mt < 8; ++mt) {
#pragma unroll
    for (int r = 0; r < 8; ++r) {
      float v = rmax[mt][r];
#pragma unroll
      for (int off = 1; off < 16; off <<= 1) v = fmaxf(v, __shfl_xor(v, off));
      if (lane == 0 || lane == 16) {
        int o = mt * 16 + ((lane == 0) ? 0 : 8) + r;
        outF[(size_t)(b * 128 + o) * Mc + m] = v;
      }
    }
  }
}

// ---------------------------------------------------------------------------
extern "C" void kernel_launch(void* const* d_in, const int* in_sizes, int n_in,
                              void* d_out, int out_size, void* d_ws, size_t ws_size,
                              hipStream_t stream) {
  const float* xyz  = (const float*)d_in[0];
  const float* feat = (const float*)d_in[1];
  const float* w0   = (const float*)d_in[2];
  const float* s0   = (const float*)d_in[3];
  const float* b0   = (const float*)d_in[4];
  const float* w1   = (const float*)d_in[5];
  const float* s1   = (const float*)d_in[6];
  const float* b1   = (const float*)d_in[7];
  const float* w2   = (const float*)d_in[8];
  const float* s2   = (const float*)d_in[9];
  const float* b2   = (const float*)d_in[10];

  float* outNew  = (float*)d_out;               // [B,3,M]
  float* outFeat = outNew + Bn * 3 * Mc;        // [B,128,M]

  int* cidx = (int*)d_ws;                       // [B*M]
  int* nidx = cidx + Bn * Mc;                   // [B*M*K]

  fps_kernel<<<Bn, 1024, 0, stream>>>(xyz, cidx);
  gather_newxyz_kernel<<<(Bn * Mc) / 256, 256, 0, stream>>>(xyz, cidx, outNew);
  ballq_kernel<<<(Bn * Mc) / 8, 256, 0, stream>>>(xyz, outNew, nidx);
  mlp_kernel<<<(Bn * Mc) / 4, 128, 0, stream>>>(xyz, feat, outNew, nidx,
                                                w0, s0, b0, w1, s1, b1, w2, s2, b2,
                                                outFeat);
}
